// GGNNLayer_10977936408823
// MI455X (gfx1250) — matrix operand order repaired and checked by reading, not compiled, not run on hardware
//
#include <hip/hip_runtime.h>
#include <hip/hip_bf16.h>

#define HID 256
#define HID3 768

typedef __attribute__((ext_vector_type(16))) __bf16 v16bf;
typedef __attribute__((ext_vector_type(8)))  float  v8f;

union FragU { v16bf v; uint4 u[2]; };

__device__ __forceinline__ unsigned short f2bf(float f) {
    unsigned u = __float_as_uint(f);
    u += 0x7FFFu + ((u >> 16) & 1u);          // round-to-nearest-even
    return (unsigned short)(u >> 16);
}
__device__ __forceinline__ float bf2f(unsigned short h) {
    return __uint_as_float(((unsigned)h) << 16);
}
__device__ __forceinline__ float sigmoidf(float x) {
    return 1.0f / (1.0f + __expf(-x));
}
// Guaranteed-native CDNA5 f32 atomic add (no return -> STOREcnt path).
__device__ __forceinline__ void atomic_add_f32(float* p, float v) {
    asm volatile("global_atomic_add_f32 %0, %1, off" :: "v"(p), "v"(v) : "memory");
}

// ---- WMMA fragment loaders -------------------------------------------------
// A fragment (16x32 bf16) from an LDS-staged row tile (stride HID elements).
// lane&15 = row M; lane>>4 selects K-half: chunks [k0+8h, +8) and [k0+16+8h, +8).
__device__ __forceinline__ v16bf load_a_lds(const unsigned short* s, int k0, int lane) {
    int m = lane & 15;
    int h = lane >> 4;
    const unsigned short* p = s + m * HID + k0 + 8 * h;
    FragU f;
    f.u[0] = *(const uint4*)(p);
    f.u[1] = *(const uint4*)(p + 16);
    return f.v;
}
// B fragment (32x16 bf16) from PRE-TRANSPOSED weights W[n][k] (row-major, stride=K).
// lane&15 = col N; lane>>4 selects K-half: chunks [k0+8h, +8) and [k0+16+8h, +8).
__device__ __forceinline__ v16bf load_b(const unsigned short* __restrict__ Wt,
                                        int stride, int n0, int k0, int lane) {
    int n = n0 + (lane & 15);
    int kh = lane >> 4;
    const unsigned short* p = Wt + (size_t)n * stride + k0 + 8 * kh;
    FragU f;
    f.u[0] = *(const uint4*)(p);
    f.u[1] = *(const uint4*)(p + 16);
    return f.v;
}
__device__ __forceinline__ v8f wmma_bf16(v16bf a, v16bf b, v8f c) {
    return __builtin_amdgcn_wmma_f32_16x16x32_bf16(false, a, false, b,
                                                   (short)0, c, false, false);
}
// Stage a 16xHID bf16 row tile into LDS (block of 256 threads, 512 uint4s).
__device__ __forceinline__ void stage_tile(unsigned short* sdst,
                                           const unsigned short* __restrict__ src,
                                           int m0, int N) {
    int tid = threadIdx.x;
#pragma unroll
    for (int it = 0; it < 2; ++it) {
        int idx = tid + it * 256;              // 0..511 uint4 chunks
        int row = idx >> 5;                    // /32 (HID/8 = 32 uint4 per row)
        int c8  = idx & 31;
        int rg  = m0 + row; if (rg >= N) rg = N - 1;
        ((uint4*)sdst)[idx] = *(const uint4*)(src + (size_t)rg * HID + c8 * 8);
    }
}

// ---------------------------------------------------------------- utilities
__global__ void k_zero_f32x4(float4* __restrict__ p, size_t n4) {
    size_t i = (size_t)blockIdx.x * blockDim.x + threadIdx.x;
    if (i < n4) p[i] = make_float4(0.f, 0.f, 0.f, 0.f);
}

__global__ void k_cast_bf16x8(unsigned short* __restrict__ dst,
                              const float* __restrict__ src, size_t n8) {
    size_t i = (size_t)blockIdx.x * blockDim.x + threadIdx.x;
    if (i >= n8) return;
    const float4* s = (const float4*)src + 2 * i;
    float4 a = s[0], b = s[1];
    union { uint4 u; unsigned short h[8]; } o;
    o.h[0] = f2bf(a.x); o.h[1] = f2bf(a.y); o.h[2] = f2bf(a.z); o.h[3] = f2bf(a.w);
    o.h[4] = f2bf(b.x); o.h[5] = f2bf(b.y); o.h[6] = f2bf(b.z); o.h[7] = f2bf(b.w);
    ((uint4*)dst)[i] = o.u;
}

// dst[b][c][r] = bf16(src[b][r][c])   (weight transpose + cast, once per launch)
__global__ void k_cvt_transpose(unsigned short* __restrict__ dst,
                                const float* __restrict__ src,
                                int B, int R, int C) {
    size_t i = (size_t)blockIdx.x * blockDim.x + threadIdx.x;
    size_t total = (size_t)B * R * C;
    if (i >= total) return;
    int b = (int)(i / ((size_t)R * C));
    int rem = (int)(i % ((size_t)R * C));
    int c = rem / R;
    int r = rem % R;
    dst[i] = f2bf(src[(size_t)b * R * C + (size_t)r * C + c]);
}

// ------------------------------------------------- per-type node transform
// out[t][n][g] = bf16( h_bf16 @ W_t + b_t ).  grid (N/16, 2, 4), block 256.
// A row tile staged in LDS; each of 8 waves computes one 16x16 column tile.
__global__ void k_transform_gemm(unsigned short* __restrict__ out,
                                 const unsigned short* __restrict__ A,
                                 const unsigned short* __restrict__ Wt,
                                 const float* __restrict__ bias, int N) {
    __shared__ __align__(16) unsigned short sA[16 * HID];
    int lane = threadIdx.x & 31;
    int wave = threadIdx.x >> 5;
    int t = blockIdx.z;
    int n0 = (blockIdx.y * 8 + wave) * 16;
    int m0 = blockIdx.x * 16;
    const unsigned short* W = Wt + (size_t)t * HID * HID;

    stage_tile(sA, A, m0, N);
    __syncthreads();

    v8f acc = {};
#pragma unroll
    for (int k0 = 0; k0 < HID; k0 += 32) {
        v16bf a = load_a_lds(sA, k0, lane);
        v16bf b = load_b(W, HID, n0, k0, lane);
        acc = wmma_bf16(a, b, acc);
    }
    int n = n0 + (lane & 15);
    int hi = lane >> 4;
    float bv = bias[t * HID + n];
    unsigned short* orow = out + ((size_t)t * N + m0 + 8 * hi) * HID + n;
#pragma unroll
    for (int r = 0; r < 8; ++r)
        if (m0 + 8 * hi + r < N)
            orow[(size_t)r * HID] = f2bf(acc[r] + bv);
}

// ----------------------------------------- edge gather + atomic scatter-add
// one wave per edge: lane loads 8 bf16 (b128) of transformed[type][src],
// native-f32-atomic-adds into messages[tgt].
__global__ void k_gather_scatter(float* __restrict__ msgs,
                                 const unsigned short* __restrict__ T,
                                 const int* __restrict__ et,
                                 const int* __restrict__ es,
                                 const int* __restrict__ eg,
                                 int E, int N) {
    int wave = threadIdx.x >> 5;
    int lane = threadIdx.x & 31;
    int e = blockIdx.x * 8 + wave;
    if (e >= E) return;
    int t = et[e], s = es[e], g = eg[e];
    const uint4 pk = *(const uint4*)(T + ((size_t)t * N + s) * HID + lane * 8);
    float* dst = msgs + (size_t)g * HID + lane * 8;
    const unsigned int* w = (const unsigned int*)&pk;
#pragma unroll
    for (int j = 0; j < 4; ++j) {
        unsigned int v = w[j];
        atomic_add_f32(dst + 2 * j,     bf2f((unsigned short)(v & 0xffffu)));
        atomic_add_f32(dst + 2 * j + 1, bf2f((unsigned short)(v >> 16)));
    }
}

// --------------------------------------------------------- fused GRU cell
// grid (N/16, 2), block 256.  Both A row tiles (messages, h) staged in LDS;
// each wave computes one 16-col output tile: 6 gate tiles (48 WMMAs), then
// the sigmoid/tanh blend, writing h_next in f32.
__global__ void k_gru_fused(float* __restrict__ hout,
                            const unsigned short* __restrict__ Mb,   // messages bf16 [N,256]
                            const unsigned short* __restrict__ Hb,   // h bf16 [N,256]
                            const float* __restrict__ hprev,         // h f32 [N,256]
                            const unsigned short* __restrict__ GkT,  // [768,256] bf16 B[n][k]
                            const unsigned short* __restrict__ GrT,  // [768,256] bf16
                            const float* __restrict__ gbias,         // [2,768]
                            int N) {
    __shared__ __align__(16) unsigned short sM[16 * HID];
    __shared__ __align__(16) unsigned short sH[16 * HID];
    int lane = threadIdx.x & 31;
    int wave = threadIdx.x >> 5;
    int m0 = blockIdx.x * 16;
    int c0 = (blockIdx.y * 8 + wave) * 16;

    stage_tile(sM, Mb, m0, N);
    stage_tile(sH, Hb, m0, N);
    __syncthreads();

    v8f axz = {}, axr = {}, axh = {}, ahz = {}, ahr = {}, ahh = {};
#pragma unroll
    for (int k0 = 0; k0 < HID; k0 += 32) {
        v16bf am = load_a_lds(sM, k0, lane);
        v16bf ah = load_a_lds(sH, k0, lane);
        v16bf b0 = load_b(GkT, HID, c0,       k0, lane);
        v16bf b1 = load_b(GkT, HID, 256 + c0, k0, lane);
        v16bf b2 = load_b(GkT, HID, 512 + c0, k0, lane);
        axz = wmma_bf16(am, b0, axz);
        axr = wmma_bf16(am, b1, axr);
        axh = wmma_bf16(am, b2, axh);
        b0 = load_b(GrT, HID, c0,       k0, lane);
        b1 = load_b(GrT, HID, 256 + c0, k0, lane);
        b2 = load_b(GrT, HID, 512 + c0, k0, lane);
        ahz = wmma_bf16(ah, b0, ahz);
        ahr = wmma_bf16(ah, b1, ahr);
        ahh = wmma_bf16(ah, b2, ahh);
    }

    int n = lane & 15, hi = lane >> 4;
    int c = c0 + n;
    float bxz = gbias[c],        bxr = gbias[256 + c],        bxh = gbias[512 + c];
    float bhz = gbias[HID3 + c], bhr = gbias[HID3 + 256 + c], bhh = gbias[HID3 + 512 + c];
    size_t base = (size_t)(m0 + 8 * hi) * HID + c;
#pragma unroll
    for (int r = 0; r < 8; ++r) {
        if (m0 + 8 * hi + r >= N) break;
        float z  = sigmoidf(axz[r] + bxz + ahz[r] + bhz);
        float rg = sigmoidf(axr[r] + bxr + ahr[r] + bhr);
        float hc = tanhf(axh[r] + bxh + rg * (ahh[r] + bhh));
        float hp = hprev[base + (size_t)r * HID];
        hout[base + (size_t)r * HID] = z * hp + (1.0f - z) * hc;
    }
}

// --------------------------------------------------------------- launcher
extern "C" void kernel_launch(void* const* d_in, const int* in_sizes, int n_in,
                              void* d_out, int out_size, void* d_ws, size_t ws_size,
                              hipStream_t stream) {
    const float* states = (const float*)d_in[0];
    const int*   etype  = (const int*)d_in[1];
    const int*   esrc   = (const int*)d_in[2];
    const int*   etgt   = (const int*)d_in[3];
    const float* tW     = (const float*)d_in[4];
    const float* tB     = (const float*)d_in[5];
    const float* gk     = (const float*)d_in[6];
    const float* grk    = (const float*)d_in[7];
    const float* gb     = (const float*)d_in[8];

    const int N = in_sizes[0] / HID;
    const int E = in_sizes[1];
    const int tiles = (N + 15) / 16;

    auto align256 = [](size_t x) { return (x + 255) & ~(size_t)255; };
    char* ws = (char*)d_ws;
    size_t off = 0;
    unsigned short* wT   = (unsigned short*)(ws + off); off = align256(off + (size_t)4 * HID * HID * 2);
    unsigned short* gkT  = (unsigned short*)(ws + off); off = align256(off + (size_t)HID3 * HID * 2);
    unsigned short* grT  = (unsigned short*)(ws + off); off = align256(off + (size_t)HID3 * HID * 2);
    unsigned short* hbf  = (unsigned short*)(ws + off); off = align256(off + (size_t)N * HID * 2);
    unsigned short* trn  = (unsigned short*)(ws + off); off = align256(off + (size_t)4 * N * HID * 2);
    float*          msgs = (float*)(ws + off);          off = align256(off + (size_t)N * HID * 4);
    unsigned short* mbf  = (unsigned short*)(ws + off); off = align256(off + (size_t)N * HID * 4 / 2);
    float*          h1   = (float*)(ws + off);          off = align256(off + (size_t)N * HID * 4);

    // weight prep (transpose to [n][k] + cast to bf16)
    {
        size_t t0 = (size_t)4 * HID * HID;
        k_cvt_transpose<<<(unsigned)((t0 + 255) / 256), 256, 0, stream>>>(wT, tW, 4, HID, HID);
        size_t t1 = (size_t)HID * HID3;
        k_cvt_transpose<<<(unsigned)((t1 + 255) / 256), 256, 0, stream>>>(gkT, gk, 1, HID, HID3);
        k_cvt_transpose<<<(unsigned)((t1 + 255) / 256), 256, 0, stream>>>(grT, grk, 1, HID, HID3);
    }

    const size_t nh = (size_t)N * HID;
    const float* hcur = states;
    for (int step = 0; step < 2; ++step) {
        float* hnext = (step == 0) ? h1 : (float*)d_out;

        k_cast_bf16x8<<<(unsigned)((nh / 8 + 255) / 256), 256, 0, stream>>>(hbf, hcur, nh / 8);

        k_transform_gemm<<<dim3(tiles, 2, 4), 256, 0, stream>>>(trn, hbf, wT, tB, N);

        k_zero_f32x4<<<(unsigned)((nh / 4 + 255) / 256), 256, 0, stream>>>((float4*)msgs, nh / 4);

        k_gather_scatter<<<(unsigned)((E + 7) / 8), 256, 0, stream>>>(msgs, trn, etype,
                                                                      esrc, etgt, E, N);

        k_cast_bf16x8<<<(unsigned)((nh / 8 + 255) / 256), 256, 0, stream>>>(mbf, msgs, nh / 8);

        k_gru_fused<<<dim3(tiles, 2), 256, 0, stream>>>(hnext, mbf, hbf, hcur,
                                                        gkT, grT, gb, N);
        hcur = hnext;
    }
    (void)n_in; (void)out_size; (void)ws_size;
}
